// NonLocalAttention_73744588472613
// MI455X (gfx1250) — matrix operand order repaired and checked by reading, not compiled, not run on hardware
//
#include <hip/hip_runtime.h>
#include <hip/hip_bf16.h>

// ---------------------------------------------------------------------------
// NonLocalAttention on MI455X (gfx1250): all GEMMs via v_wmma_f32_16x16x32_bf16,
// flash-style online-softmax attention (never materializes the 134MB NxN map).
// ---------------------------------------------------------------------------

typedef __bf16 bf16_t;
typedef __attribute__((ext_vector_type(8)))  __bf16 v8bf;
typedef __attribute__((ext_vector_type(16))) __bf16 v16bf;
typedef __attribute__((ext_vector_type(8)))  float  v8f;

#define Bn 2
#define Cc 256
#define CI 128
#define Nn 4096   // H*W = 64*64
#define Ll 4
#define HW 64

__device__ __forceinline__ v8f zero8f() {
    v8f z = {0.f,0.f,0.f,0.f,0.f,0.f,0.f,0.f};
    return z;
}
__device__ __forceinline__ v16bf zero16bf() {
    bf16_t z = (bf16_t)0.0f;
    v16bf v = {z,z,z,z,z,z,z,z,z,z,z,z,z,z,z,z};
    return v;
}

// A-fragment (16x32 bf16): lane holds row (caller passes row = base + (lane&15)),
// K = {kb..kb+7, 16+kb..16+kb+7}, kb = 8*(lane>>4). Two aligned 16B loads.
__device__ __forceinline__ v16bf load_a(const bf16_t* __restrict__ p, int row, int ld, int k0) {
    int lane = threadIdx.x & 31;
    int kb = (lane >> 4) << 3;
    const bf16_t* q = p + (size_t)row * ld + k0 + kb;
    v8bf lo = *reinterpret_cast<const v8bf*>(q);
    v8bf hi = *reinterpret_cast<const v8bf*>(q + 16);
    return __builtin_shufflevector(lo, hi, 0,1,2,3,4,5,6,7,8,9,10,11,12,13,14,15);
}

// B-fragment (32x16 bf16): lane holds column (caller passes row-in-storage =
// colbase + (lane&15)), K = {kb..kb+7, 16+kb..16+kb+7}, kb = 8*(lane>>4). Two 16B loads.
__device__ __forceinline__ v16bf load_b(const bf16_t* __restrict__ p, int row, int ld, int k0) {
    int lane = threadIdx.x & 31;
    int kb = (lane >> 4) << 3;
    const bf16_t* q = p + (size_t)row * ld + k0 + kb;
    v8bf lo = *reinterpret_cast<const v8bf*>(q);
    v8bf hi = *reinterpret_cast<const v8bf*>(q + 16);
    return __builtin_shufflevector(lo, hi, 0,1,2,3,4,5,6,7,8,9,10,11,12,13,14,15);
}

__device__ __forceinline__ v16bf load_b_pred(const bf16_t* __restrict__ p, int row, int ld,
                                             int k0, bool valid) {
    v16bf z = zero16bf();
    if (valid) z = load_b(p, row, ld, k0);   // diverged loads; reconverges before WMMA
    return z;
}

__device__ __forceinline__ v8f wmma_bf16(v16bf a, v16bf b, v8f c) {
    return __builtin_amdgcn_wmma_f32_16x16x32_bf16(false, a, false, b, (short)0, c, false, false);
}

// ---------------------------------------------------------------------------
// Prep kernels (run once per launch; deterministic)
// ---------------------------------------------------------------------------
__global__ void k_cvt_bf16(const float* __restrict__ src, bf16_t* __restrict__ dst, int n) {
    for (int i = blockIdx.x * blockDim.x + threadIdx.x; i < n; i += gridDim.x * blockDim.x)
        dst[i] = (bf16_t)src[i];
}

// conv_w [L,O,C,3,3] -> wpack [L][O][9][C] bf16 (k = j*C + c, contiguous in c)
__global__ void k_pack_convw(const float* __restrict__ src, bf16_t* __restrict__ dst, int total) {
    for (int i = blockIdx.x * blockDim.x + threadIdx.x; i < total; i += gridDim.x * blockDim.x) {
        int j = i % 9;
        int c = (i / 9) % Cc;
        int o = (i / (9 * Cc)) % Cc;
        int l = i / (9 * Cc * Cc);
        dst[(((size_t)l * Cc + o) * 9 + j) * Cc + c] = (bf16_t)src[i];
    }
}

__global__ void k_bn_consts(const float* __restrict__ gamma, const float* __restrict__ beta,
                            const float* __restrict__ mean,  const float* __restrict__ var,
                            float* __restrict__ s, float* __restrict__ t, int n) {
    for (int i = blockIdx.x * blockDim.x + threadIdx.x; i < n; i += gridDim.x * blockDim.x) {
        float inv = gamma[i] * rsqrtf(var[i] + 1e-5f);
        s[i] = inv;
        t[i] = beta[i] - mean[i] * inv;
    }
}

// x [B,C,N] f32 -> xT [B,N,C] bf16
__global__ void k_transpose_cvt(const float* __restrict__ x, bf16_t* __restrict__ xT, int total) {
    for (int i = blockIdx.x * blockDim.x + threadIdx.x; i < total; i += gridDim.x * blockDim.x) {
        int n = i % Nn;
        int c = (i / Nn) % Cc;
        int b = i / (Nn * Cc);
        xT[((size_t)b * Nn + n) * Cc + c] = (bf16_t)x[i];
    }
}

// ---------------------------------------------------------------------------
// Projection GEMM: out = W(Ci x C) * X(C x N) + bias, all three projections.
// grid (N/64, B, 3proj), 8 waves; wave w owns Ci-rows [16w,16w+16), 64 cols.
// ---------------------------------------------------------------------------
__global__ void __launch_bounds__(256)
k_proj(const bf16_t* __restrict__ xT,
       const bf16_t* __restrict__ wg, const bf16_t* __restrict__ wth, const bf16_t* __restrict__ wph,
       const float* __restrict__ bg,  const float* __restrict__ bth,  const float* __restrict__ bph,
       bf16_t* __restrict__ out_g,    bf16_t* __restrict__ out_thT,   bf16_t* __restrict__ out_phT) {
    int pz = blockIdx.z;
    const bf16_t* w   = (pz == 0) ? wg : (pz == 1) ? wth : wph;
    const float*  bia = (pz == 0) ? bg : (pz == 1) ? bth : bph;
    int b  = blockIdx.y;
    int n0 = blockIdx.x * 64;
    int wave = threadIdx.x >> 5, lane = threadIdx.x & 31;
    int r = lane & 15, half = lane >> 4;
    int ci0 = wave * 16;
    const bf16_t* xb = xT + (size_t)b * Nn * Cc;

    v8f acc[4] = {zero8f(), zero8f(), zero8f(), zero8f()};
    for (int k0 = 0; k0 < Cc; k0 += 32) {
        v16bf a = load_a(w, ci0 + r, Cc, k0);
#pragma unroll
        for (int t = 0; t < 4; ++t) {
            v16bf bb = load_b(xb, n0 + t * 16 + r, Cc, k0);
            acc[t] = wmma_bf16(a, bb, acc[t]);
        }
    }
#pragma unroll
    for (int t = 0; t < 4; ++t)
#pragma unroll
        for (int rr = 0; rr < 8; ++rr) {
            int ci = ci0 + rr + 8 * half;  // D row
            int n  = n0 + t * 16 + r;      // D col
            float v = acc[t][rr] + bia[ci];
            if (pz == 0)      out_g  [((size_t)b * CI + ci) * Nn + n] = (bf16_t)v; // [Ci,N]
            else if (pz == 1) out_thT[((size_t)b * Nn + n) * CI + ci] = (bf16_t)v; // [N,Ci]
            else              out_phT[((size_t)b * Nn + n) * CI + ci] = (bf16_t)v; // [N,Ci]
        }
}

// ---------------------------------------------------------------------------
// Flash attention: per wave, one 16-row query tile; theta A-frags resident in
// VGPRs; online softmax with half-wave shuffle reductions; exp(P) staged
// through LDS to convert C-fragment layout -> A-fragment layout.
// grid (N/128, B), 8 waves.
// ---------------------------------------------------------------------------
__global__ void __launch_bounds__(256)
k_attn(const bf16_t* __restrict__ thT, const bf16_t* __restrict__ phT,
       const bf16_t* __restrict__ g,   bf16_t* __restrict__ yT) {
    __shared__ __align__(16) bf16_t lds[8][16][32];
    int b = blockIdx.y;
    int wave = threadIdx.x >> 5, lane = threadIdx.x & 31;
    int r = lane & 15, half = lane >> 4;
    int q0 = (blockIdx.x * 8 + wave) * 16;
    const bf16_t* thb = thT + (size_t)b * Nn * CI;
    const bf16_t* phb = phT + (size_t)b * Nn * CI;
    const bf16_t* gb  = g   + (size_t)b * CI * Nn;

    v16bf aq[4];
#pragma unroll
    for (int kc = 0; kc < 4; ++kc) aq[kc] = load_a(thb, q0 + r, CI, kc * 32);

    v8f y[8];
    float rmax[8], rsum[8];
#pragma unroll
    for (int t = 0; t < 8; ++t) y[t] = zero8f();
#pragma unroll
    for (int i = 0; i < 8; ++i) { rmax[i] = -1e30f; rsum[i] = 0.f; }

    for (int m0 = 0; m0 < Nn; m0 += 32) {
        v8f s0 = zero8f(), s1 = zero8f();
#pragma unroll
        for (int kc = 0; kc < 4; ++kc) {
            v16bf b0 = load_b(phb, m0 + r,      CI, kc * 32);
            v16bf b1 = load_b(phb, m0 + 16 + r, CI, kc * 32);
            s0 = wmma_bf16(aq[kc], b0, s0);
            s1 = wmma_bf16(aq[kc], b1, s1);
        }
        float corr[8];
#pragma unroll
        for (int rr = 0; rr < 8; ++rr) {
            float m = fmaxf(s0[rr], s1[rr]);
            m = fmaxf(m, __shfl_xor(m, 1));
            m = fmaxf(m, __shfl_xor(m, 2));
            m = fmaxf(m, __shfl_xor(m, 4));
            m = fmaxf(m, __shfl_xor(m, 8));   // half-wave (16-lane) row max
            float nm = fmaxf(rmax[rr], m);
            corr[rr] = __expf(rmax[rr] - nm);
            rmax[rr] = nm;
            float e0 = __expf(s0[rr] - nm);
            float e1 = __expf(s1[rr] - nm);
            float ps = e0 + e1;
            ps += __shfl_xor(ps, 1);
            ps += __shfl_xor(ps, 2);
            ps += __shfl_xor(ps, 4);
            ps += __shfl_xor(ps, 8);
            rsum[rr] = rsum[rr] * corr[rr] + ps;
            lds[wave][rr + 8 * half][r]      = (bf16_t)e0;
            lds[wave][rr + 8 * half][r + 16] = (bf16_t)e1;
        }
#pragma unroll
        for (int t = 0; t < 8; ++t)
#pragma unroll
            for (int rr = 0; rr < 8; ++rr) y[t][rr] *= corr[rr];

        // re-read P as an A-fragment (same-wave LDS ops are in-order)
        v16bf pA;
        {
            const bf16_t* q = &lds[wave][r][half << 3];
            v8bf lo = *reinterpret_cast<const v8bf*>(q);
            v8bf hi = *reinterpret_cast<const v8bf*>(q + 16);
            pA = __builtin_shufflevector(lo, hi, 0,1,2,3,4,5,6,7,8,9,10,11,12,13,14,15);
        }
#pragma unroll
        for (int t = 0; t < 8; ++t) {
            v16bf gB = load_b(gb, t * 16 + r, Nn, m0);  // G[k=m][n=ci] = g[ci][m]
            y[t] = wmma_bf16(pA, gB, y[t]);
        }
    }
#pragma unroll
    for (int t = 0; t < 8; ++t)
#pragma unroll
        for (int rr = 0; rr < 8; ++rr) {
            int n  = q0 + rr + 8 * half;
            int ci = t * 16 + r;
            yT[((size_t)b * Nn + n) * CI + ci] = (bf16_t)(y[t][rr] / rsum[rr]);
        }
}

// ---------------------------------------------------------------------------
// Wy = W_w(C x Ci) * y(Ci x N); then BN1 + residual; emit x1T [B,N,C] bf16.
// grid (N/64, B, 2), 8 waves; wave owns C-rows [16*(8z+w), +16).
// ---------------------------------------------------------------------------
__global__ void __launch_bounds__(256)
k_wgemm(const bf16_t* __restrict__ yT, const bf16_t* __restrict__ Ww,
        const float* __restrict__ Wb, const float* __restrict__ s1, const float* __restrict__ t1,
        const float* __restrict__ xin, bf16_t* __restrict__ x1T) {
    int b  = blockIdx.y;
    int n0 = blockIdx.x * 64;
    int wave = threadIdx.x >> 5, lane = threadIdx.x & 31;
    int r = lane & 15, half = lane >> 4;
    int c0 = (blockIdx.z * 8 + wave) * 16;
    const bf16_t* yb = yT + (size_t)b * Nn * CI;

    v8f acc[4] = {zero8f(), zero8f(), zero8f(), zero8f()};
#pragma unroll
    for (int k0 = 0; k0 < CI; k0 += 32) {
        v16bf a = load_a(Ww, c0 + r, CI, k0);
#pragma unroll
        for (int t = 0; t < 4; ++t) {
            v16bf bb = load_b(yb, n0 + t * 16 + r, CI, k0);
            acc[t] = wmma_bf16(a, bb, acc[t]);
        }
    }
    const float* xb = xin + (size_t)b * Cc * Nn;
#pragma unroll
    for (int t = 0; t < 4; ++t)
#pragma unroll
        for (int rr = 0; rr < 8; ++rr) {
            int c = c0 + rr + 8 * half;
            int n = n0 + t * 16 + r;
            float v = acc[t][rr] + Wb[c];
            v = v * s1[c] + t1[c];
            v += xb[(size_t)c * Nn + n];
            x1T[((size_t)b * Nn + n) * Cc + c] = (bf16_t)v;
        }
}

// ---------------------------------------------------------------------------
// 3x3 conv as implicit GEMM (9 shifted K=256 GEMMs) + BN2 + ReLU -> f32 out.
// grid (N/64, B, 2), 8 waves.
// ---------------------------------------------------------------------------
__global__ void __launch_bounds__(256)
k_conv(const bf16_t* __restrict__ x1T, const bf16_t* __restrict__ wp /*[O][9][C]*/,
       const float* __restrict__ cb, const float* __restrict__ s2, const float* __restrict__ t2,
       float* __restrict__ out) {
    int b  = blockIdx.y;
    int n0 = blockIdx.x * 64;
    int wave = threadIdx.x >> 5, lane = threadIdx.x & 31;
    int r = lane & 15, half = lane >> 4;
    int o0 = (blockIdx.z * 8 + wave) * 16;
    const bf16_t* xb = x1T + (size_t)b * Nn * Cc;

    v8f acc[4] = {zero8f(), zero8f(), zero8f(), zero8f()};
    for (int j = 0; j < 9; ++j) {
        int dy = j / 3 - 1, dx = j % 3 - 1;
        int  pr[4];
        bool pv[4];
#pragma unroll
        for (int t = 0; t < 4; ++t) {
            int p  = n0 + t * 16 + r;
            int px = (p & (HW - 1)) + dx;
            int py = (p >> 6) + dy;
            pv[t] = (px >= 0) & (px < HW) & (py >= 0) & (py < HW);
            pr[t] = py * HW + px;
        }
        for (int cc = 0; cc < Cc; cc += 32) {
            v16bf a = load_a(wp, o0 + r, 9 * Cc, j * Cc + cc);
#pragma unroll
            for (int t = 0; t < 4; ++t) {
                v16bf bb = load_b_pred(xb, pr[t], Cc, cc, pv[t]);
                acc[t] = wmma_bf16(a, bb, acc[t]);
            }
        }
    }
#pragma unroll
    for (int t = 0; t < 4; ++t)
#pragma unroll
        for (int rr = 0; rr < 8; ++rr) {
            int o = o0 + rr + 8 * half;
            int p = n0 + t * 16 + r;
            float v = acc[t][rr] + cb[o];
            v = fmaxf(v * s2[o] + t2[o], 0.f);
            out[((size_t)b * Cc + o) * Nn + p] = v;
        }
}

// ---------------------------------------------------------------------------
// Host launcher
// ---------------------------------------------------------------------------
extern "C" void kernel_launch(void* const* d_in, const int* in_sizes, int n_in,
                              void* d_out, int out_size, void* d_ws, size_t ws_size,
                              hipStream_t stream) {
    (void)in_sizes; (void)n_in; (void)out_size; (void)ws_size;
    const float* x_in  = (const float*)d_in[0];
    const float* g_w   = (const float*)d_in[1];
    const float* g_b   = (const float*)d_in[2];
    const float* th_w  = (const float*)d_in[3];
    const float* th_b  = (const float*)d_in[4];
    const float* ph_w  = (const float*)d_in[5];
    const float* ph_b  = (const float*)d_in[6];
    const float* W_w   = (const float*)d_in[7];
    const float* W_b   = (const float*)d_in[8];
    const float* bn1g  = (const float*)d_in[9];
    const float* bn1b  = (const float*)d_in[10];
    const float* bn1m  = (const float*)d_in[11];
    const float* bn1v  = (const float*)d_in[12];
    const float* convw = (const float*)d_in[13];
    const float* convb = (const float*)d_in[14];
    const float* bn2g  = (const float*)d_in[15];
    const float* bn2b  = (const float*)d_in[16];
    const float* bn2m  = (const float*)d_in[17];
    const float* bn2v  = (const float*)d_in[18];

    // workspace carve-up (256B aligned)
    char* base = (char*)d_ws;
    size_t off = 0;
    auto alloc = [&](size_t bytes) -> void* {
        off = (off + 255) & ~(size_t)255;
        void* p = base + off;
        off += bytes;
        return p;
    };
    const size_t projW = (size_t)Ll * CI * Cc;
    bf16_t* gwb   = (bf16_t*)alloc(projW * 2);
    bf16_t* thwb  = (bf16_t*)alloc(projW * 2);
    bf16_t* phwb  = (bf16_t*)alloc(projW * 2);
    bf16_t* Wwb   = (bf16_t*)alloc((size_t)Ll * Cc * CI * 2);
    bf16_t* wpack = (bf16_t*)alloc((size_t)Ll * Cc * 9 * Cc * 2);
    float*  s1    = (float*) alloc((size_t)Ll * Cc * 4);
    float*  t1    = (float*) alloc((size_t)Ll * Cc * 4);
    float*  s2    = (float*) alloc((size_t)Ll * Cc * 4);
    float*  t2    = (float*) alloc((size_t)Ll * Cc * 4);
    bf16_t* xT    = (bf16_t*)alloc((size_t)Bn * Nn * Cc * 2);
    bf16_t* gbuf  = (bf16_t*)alloc((size_t)Bn * CI * Nn * 2);
    bf16_t* thT   = (bf16_t*)alloc((size_t)Bn * Nn * CI * 2);
    bf16_t* phT   = (bf16_t*)alloc((size_t)Bn * Nn * CI * 2);
    bf16_t* yT    = (bf16_t*)alloc((size_t)Bn * Nn * CI * 2);
    bf16_t* x1T   = (bf16_t*)alloc((size_t)Bn * Nn * Cc * 2);
    float*  xbufA = (float*) alloc((size_t)Bn * Cc * Nn * 4);
    float*  xbufB = (float*) alloc((size_t)Bn * Cc * Nn * 4);

    // one-time (per launch) weight conversion / packing / BN folding
    k_cvt_bf16 <<<256, 256, 0, stream>>>(g_w,  gwb,  (int)projW);
    k_cvt_bf16 <<<256, 256, 0, stream>>>(th_w, thwb, (int)projW);
    k_cvt_bf16 <<<256, 256, 0, stream>>>(ph_w, phwb, (int)projW);
    k_cvt_bf16 <<<256, 256, 0, stream>>>(W_w,  Wwb,  Ll * Cc * CI);
    k_pack_convw<<<2048, 256, 0, stream>>>(convw, wpack, Ll * Cc * Cc * 9);
    k_bn_consts<<<4, 256, 0, stream>>>(bn1g, bn1b, bn1m, bn1v, s1, t1, Ll * Cc);
    k_bn_consts<<<4, 256, 0, stream>>>(bn2g, bn2b, bn2m, bn2v, s2, t2, Ll * Cc);

    const float* xin = x_in;
    for (int l = 0; l < Ll; ++l) {
        float* xout = (l == Ll - 1) ? (float*)d_out : ((l & 1) ? xbufB : xbufA);

        k_transpose_cvt<<<2048, 256, 0, stream>>>(xin, xT, Bn * Cc * Nn);

        k_proj<<<dim3(Nn / 64, Bn, 3), 256, 0, stream>>>(
            xT,
            gwb + (size_t)l * CI * Cc, thwb + (size_t)l * CI * Cc, phwb + (size_t)l * CI * Cc,
            g_b + (size_t)l * CI, th_b + (size_t)l * CI, ph_b + (size_t)l * CI,
            gbuf, thT, phT);

        k_attn<<<dim3(Nn / 128, Bn), 256, 0, stream>>>(thT, phT, gbuf, yT);

        k_wgemm<<<dim3(Nn / 64, Bn, 2), 256, 0, stream>>>(
            yT, Wwb + (size_t)l * Cc * CI, W_b + (size_t)l * Cc,
            s1 + (size_t)l * Cc, t1 + (size_t)l * Cc, xin, x1T);

        k_conv<<<dim3(Nn / 64, Bn, 2), 256, 0, stream>>>(
            x1T, wpack + (size_t)l * Cc * 9 * Cc, convb + (size_t)l * Cc,
            s2 + (size_t)l * Cc, t2 + (size_t)l * Cc, xout);

        xin = xout;
    }
}
